// VGT_41248865910972
// MI455X (gfx1250) — matrix operand order repaired and checked by reading, not compiled, not run on hardware
//
#include <hip/hip_runtime.h>
#include <hip/hip_bf16.h>

// ---------------------------------------------------------------------------
// Types for CDNA5 WMMA (wave32)
// ---------------------------------------------------------------------------
typedef __attribute__((ext_vector_type(16))) __bf16 v16bf;
typedef __attribute__((ext_vector_type(8)))  __bf16 v8bf;
typedef __attribute__((ext_vector_type(8)))  float  v8f;

// Model constants (match reference)
#define MD   1024      // d_model
#define MH   16        // heads
#define MDH  64        // head dim
#define MF   4096      // ffn hidden
#define MB   2         // batch
#define MS   1024      // seq len
#define MM   (MB*MS)   // total rows = 2048
#define ML   4         // layers

// ---------------------------------------------------------------------------
// Generic batched WMMA GEMM:  C[z] = act( (A[z] * B[z]^T_stored + bias) * alpha )
//   A  : M x K, bf16, row-major, leading dim lda, batch offset (z/zdiv)*a_s1+(z%zdiv)*a_s2
//   Bt : N x K, bf16, row-major (i.e. B stored transposed), leading dim ldb
//   out: M x N row-major (ldo), fp32 (outF) and/or bf16 (outB), same offsets
// Block: 128 threads = 4 waves in 2x2; block tile 128x64; wave tile 64x32
// (4x2 WMMA 16x16x32_bf16 accumulators). Double-buffered fragments (ping/pong)
// so prefetch loads never overwrite WMMA source VGPRs (avoids WAR hazard NOPs).
// Requires M%128==0, N%64==0, K%64==0.
// ---------------------------------------------------------------------------
__global__ void __launch_bounds__(128)
wmma_gemm(const __bf16* __restrict__ A, long a_s1, long a_s2, int lda,
          const __bf16* __restrict__ Bt, long b_s1, long b_s2, int ldb,
          float* outF, __bf16* outB, long o_s1, long o_s2, int ldo,
          const float* __restrict__ bias, int K, float alpha, int zdiv, int gelu)
{
    const int z     = blockIdx.z;
    const long aoff = (long)(z / zdiv) * a_s1 + (long)(z % zdiv) * a_s2;
    const long boff = (long)(z / zdiv) * b_s1 + (long)(z % zdiv) * b_s2;
    const long ooff = (long)(z / zdiv) * o_s1 + (long)(z % zdiv) * o_s2;

    const int lane = threadIdx.x & 31;
    const int wave = threadIdx.x >> 5;
    const int wm   = (wave >> 1) << 6;   // wave row offset in block tile (0/64)
    const int wn   = (wave & 1) << 5;    // wave col offset in block tile (0/32)
    const int bm   = blockIdx.y << 7;    // block tile 128 rows
    const int bn   = blockIdx.x << 6;    // block tile 64 cols
    const int r16  = lane & 15;
    const int kh   = lane >> 4;          // which K-half this lane holds

    // Per-lane row base pointers (K-offset added per step)
    const __bf16* arow[4];
    const __bf16* brow[2];
#pragma unroll
    for (int t = 0; t < 4; ++t)
        arow[t] = A + aoff + (long)(bm + wm + t * 16 + r16) * lda + kh * 8;
#pragma unroll
    for (int t = 0; t < 2; ++t)
        brow[t] = Bt + boff + (long)(bn + wn + t * 16 + r16) * ldb + kh * 8;

    // Fragment loader:
    //  A (16x32 bf16): elements 0..7 -> k0+kh*8+0..7, 8..15 -> k0+16+kh*8+0..7
    //  B (32x16 bf16): elements 0..7 -> k0+kh*8+0..7, 8..15 -> k0+16+kh*8+0..7
    auto lf = [&](int k0, v16bf af[4], v16bf bfm[2]) {
#pragma unroll
        for (int t = 0; t < 4; ++t) {
            const __bf16* ap = arow[t] + k0;
            v8bf alo = *(const v8bf*)ap;
            v8bf ahi = *(const v8bf*)(ap + 16);
#pragma unroll
            for (int i = 0; i < 8; ++i) { af[t][i] = alo[i]; af[t][i + 8] = ahi[i]; }
        }
#pragma unroll
        for (int t = 0; t < 2; ++t) {
            const __bf16* bp = brow[t] + k0;
            v8bf blo = *(const v8bf*)bp;
            v8bf bhi = *(const v8bf*)(bp + 16);
#pragma unroll
            for (int i = 0; i < 8; ++i) { bfm[t][i] = blo[i]; bfm[t][i + 8] = bhi[i]; }
        }
    };

    v8f acc[4][2];
#pragma unroll
    for (int mt = 0; mt < 4; ++mt)
#pragma unroll
        for (int nt = 0; nt < 2; ++nt)
            acc[mt][nt] = (v8f){0.f,0.f,0.f,0.f,0.f,0.f,0.f,0.f};

    auto dowmma = [&](v16bf af[4], v16bf bfm[2]) {
#pragma unroll
        for (int mt = 0; mt < 4; ++mt)
#pragma unroll
            for (int nt = 0; nt < 2; ++nt)
                acc[mt][nt] = __builtin_amdgcn_wmma_f32_16x16x32_bf16(
                    false, af[mt], false, bfm[nt], (short)0, acc[mt][nt], false, false);
    };

    // Double-buffered main loop (K % 64 == 0)
    v16bf a0[4], b0[2], a1[4], b1[2];
    lf(0, a0, b0);
    for (int k0 = 0; k0 < K; k0 += 64) {
        lf(k0 + 32, a1, b1);          // prefetch into pong while ping computes
        dowmma(a0, b0);
        int kn = k0 + 64;
        if (kn >= K) kn = 0;          // harmless redundant load on last iter
        lf(kn, a0, b0);               // prefetch into ping while pong computes
        dowmma(a1, b1);
    }

    // ---- Epilogue (C/D layout: VGPR r -> row r + 8*kh, lane -> col r16) ----
    float bv[2] = {0.f, 0.f};
    if (bias) {
#pragma unroll
        for (int nt = 0; nt < 2; ++nt)
            bv[nt] = bias[bn + wn + nt * 16 + r16];
    }
    if (gelu) {
#pragma unroll
        for (int mt = 0; mt < 4; ++mt)
#pragma unroll
            for (int nt = 0; nt < 2; ++nt)
#pragma unroll
                for (int r = 0; r < 8; ++r) {
                    float v = (acc[mt][nt][r] + bv[nt]) * alpha;
                    acc[mt][nt][r] = 0.5f * v * (1.0f + erff(v * 0.70710678118654752f));
                }
    } else {
#pragma unroll
        for (int mt = 0; mt < 4; ++mt)
#pragma unroll
            for (int nt = 0; nt < 2; ++nt)
#pragma unroll
                for (int r = 0; r < 8; ++r)
                    acc[mt][nt][r] = (acc[mt][nt][r] + bv[nt]) * alpha;
    }

    if (outF) {
#pragma unroll
        for (int mt = 0; mt < 4; ++mt)
#pragma unroll
            for (int nt = 0; nt < 2; ++nt)
#pragma unroll
                for (int r = 0; r < 8; ++r) {
                    int m = bm + wm + mt * 16 + r + 8 * kh;
                    int n = bn + wn + nt * 16 + r16;
                    outF[ooff + (long)m * ldo + n] = acc[mt][nt][r];
                }
    }
    if (outB) {
#pragma unroll
        for (int mt = 0; mt < 4; ++mt)
#pragma unroll
            for (int nt = 0; nt < 2; ++nt)
#pragma unroll
                for (int r = 0; r < 8; ++r) {
                    int m = bm + wm + mt * 16 + r + 8 * kh;
                    int n = bn + wn + nt * 16 + r16;
                    outB[ooff + (long)m * ldo + n] = (__bf16)acc[mt][nt][r];
                }
    }
}

// ---------------------------------------------------------------------------
// fp32 -> bf16 elementwise convert
// ---------------------------------------------------------------------------
__global__ void cvt_f32_bf16(const float* __restrict__ src, __bf16* __restrict__ dst, int n)
{
    int i = blockIdx.x * 256 + threadIdx.x;
    if (i < n) dst[i] = (__bf16)src[i];
}

// ---------------------------------------------------------------------------
// Transpose + convert: src (K x N fp32, row-major) -> dst (N x K bf16, row-major)
// Grid: (N/32, K/32), block (32, 8)
// ---------------------------------------------------------------------------
__global__ void transpose_cvt(const float* __restrict__ src, __bf16* __restrict__ dst,
                              int K, int N)
{
    __shared__ float t[32][33];
    const int n0 = blockIdx.x * 32;
    const int k0 = blockIdx.y * 32;
#pragma unroll
    for (int i = 0; i < 4; ++i) {
        int k = k0 + threadIdx.y + i * 8;
        int n = n0 + threadIdx.x;
        t[threadIdx.y + i * 8][threadIdx.x] = src[(long)k * N + n];
    }
    __syncthreads();
#pragma unroll
    for (int i = 0; i < 4; ++i) {
        int n = n0 + threadIdx.y + i * 8;
        int k = k0 + threadIdx.x;
        dst[(long)n * K + k] = (__bf16)t[threadIdx.x][threadIdx.y + i * 8];
    }
}

// ---------------------------------------------------------------------------
// Per-head V transpose: vb (B*S, D) bf16  ->  vt (B, H, DH, S) bf16
// idx = ((b*H + h)*DH + d)*S + j   reads  vb[(b*S + j)*D + h*DH + d]
// ---------------------------------------------------------------------------
__global__ void v_transpose(const __bf16* __restrict__ vb, __bf16* __restrict__ vt, int total)
{
    int idx = blockIdx.x * 256 + threadIdx.x;
    if (idx >= total) return;
    int j = idx & (MS - 1);
    int t = idx >> 10;
    int d = t & (MDH - 1); t >>= 6;
    int h = t & (MH - 1);
    int b = t >> 4;
    vt[idx] = vb[((long)(b * MS + j) << 10) + h * MDH + d];
}

// ---------------------------------------------------------------------------
// In-place masked row softmax on bf16 scores. One 256-thread block per row,
// 4 columns per thread (S = 1024). rows = B*H*S; b = row / (H*S).
// ---------------------------------------------------------------------------
__global__ void __launch_bounds__(256)
softmax_rows(__bf16* __restrict__ probs, const int* __restrict__ mask, int S, int HS)
{
    __shared__ float red[256];
    const long row = blockIdx.x;
    const int  b   = (int)(row / HS);
    __bf16* p = probs + row * (long)S;
    const int* mrow = mask + (long)b * S;
    const int tid = threadIdx.x;

    float x[4];
    float mx = -3.0e38f;
#pragma unroll
    for (int i = 0; i < 4; ++i) {
        int c = tid + i * 256;
        float v = (float)p[c];
        if (mrow[c] == 0) v = -3.0e38f;
        x[i] = v;
        mx = fmaxf(mx, v);
    }
    red[tid] = mx;
    __syncthreads();
    for (int st = 128; st > 0; st >>= 1) {
        if (tid < st) red[tid] = fmaxf(red[tid], red[tid + st]);
        __syncthreads();
    }
    mx = red[0];
    __syncthreads();

    float e[4];
    float sum = 0.f;
#pragma unroll
    for (int i = 0; i < 4; ++i) {
        e[i] = __expf(x[i] - mx);
        sum += e[i];
    }
    red[tid] = sum;
    __syncthreads();
    for (int st = 128; st > 0; st >>= 1) {
        if (tid < st) red[tid] += red[tid + st];
        __syncthreads();
    }
    float inv = 1.0f / red[0];
#pragma unroll
    for (int i = 0; i < 4; ++i)
        p[tid + i * 256] = (__bf16)(e[i] * inv);
}

// ---------------------------------------------------------------------------
// LayerNorm(x + res) with gamma/beta; writes fp32 (residual stream) and bf16
// (GEMM stream). One 256-thread block per row of D = 1024 (4 cols/thread).
// outF may alias res (thread reads its own elements before writing).
// ---------------------------------------------------------------------------
__global__ void __launch_bounds__(256)
ln_residual(const float* __restrict__ x, const float* res,
            const float* __restrict__ g, const float* __restrict__ be,
            float* outF, __bf16* __restrict__ outB, int D)
{
    __shared__ float red[256];
    const long row = blockIdx.x;
    const float* xr = x + row * (long)D;
    const float* rr = res + row * (long)D;
    const int tid = threadIdx.x;

    float v[4];
    float s = 0.f;
#pragma unroll
    for (int i = 0; i < 4; ++i) {
        int c = tid + i * 256;
        v[i] = xr[c] + rr[c];
        s += v[i];
    }
    red[tid] = s;
    __syncthreads();
    for (int st = 128; st > 0; st >>= 1) {
        if (tid < st) red[tid] += red[tid + st];
        __syncthreads();
    }
    const float mean = red[0] / (float)D;
    __syncthreads();

    float s2 = 0.f;
#pragma unroll
    for (int i = 0; i < 4; ++i) {
        float d = v[i] - mean;
        s2 += d * d;
    }
    red[tid] = s2;
    __syncthreads();
    for (int st = 128; st > 0; st >>= 1) {
        if (tid < st) red[tid] += red[tid + st];
        __syncthreads();
    }
    const float var  = red[0] / (float)D;
    const float rstd = rsqrtf(var + 1e-12f);

#pragma unroll
    for (int i = 0; i < 4; ++i) {
        int c = tid + i * 256;
        float y = (v[i] - mean) * rstd * g[c] + be[c];
        outF[row * (long)D + c] = y;
        outB[row * (long)D + c] = (__bf16)y;
    }
}

// ---------------------------------------------------------------------------
// Host orchestration
// ---------------------------------------------------------------------------
extern "C" void kernel_launch(void* const* d_in, const int* in_sizes, int n_in,
                              void* d_out, int out_size, void* d_ws, size_t ws_size,
                              hipStream_t stream)
{
    (void)in_sizes; (void)n_in; (void)out_size; (void)ws_size;

    const float* x    = (const float*)d_in[0];
    const int*   amsk = (const int*)  d_in[1];
    const float* Wq = (const float*)d_in[2];  const float* bq = (const float*)d_in[3];
    const float* Wk = (const float*)d_in[4];  const float* bk = (const float*)d_in[5];
    const float* Wv = (const float*)d_in[6];  const float* bv = (const float*)d_in[7];
    const float* Wo = (const float*)d_in[8];  const float* bo = (const float*)d_in[9];
    const float* g1 = (const float*)d_in[10]; const float* be1 = (const float*)d_in[11];
    const float* W1 = (const float*)d_in[12]; const float* b1 = (const float*)d_in[13];
    const float* W2 = (const float*)d_in[14]; const float* b2 = (const float*)d_in[15];
    const float* g2 = (const float*)d_in[16]; const float* be2 = (const float*)d_in[17];

    // Workspace carve-out (all 256B-aligned)
    char* wsp = (char*)d_ws;
    auto alloc = [&](size_t bytes) -> void* {
        void* p = (void*)wsp;
        wsp += (bytes + 255) & ~(size_t)255;
        return p;
    };
    float*  h     = (float*) alloc((size_t)MM * MD * 4);
    __bf16* hb    = (__bf16*)alloc((size_t)MM * MD * 2);
    __bf16* wqt   = (__bf16*)alloc((size_t)MD * MD * 2);
    __bf16* wkt   = (__bf16*)alloc((size_t)MD * MD * 2);
    __bf16* wvt   = (__bf16*)alloc((size_t)MD * MD * 2);
    __bf16* wot   = (__bf16*)alloc((size_t)MD * MD * 2);
    __bf16* w1t   = (__bf16*)alloc((size_t)MD * MF * 2);
    __bf16* w2t   = (__bf16*)alloc((size_t)MF * MD * 2);
    __bf16* qb    = (__bf16*)alloc((size_t)MM * MD * 2);
    __bf16* kb    = (__bf16*)alloc((size_t)MM * MD * 2);
    __bf16* vb    = (__bf16*)alloc((size_t)MM * MD * 2);
    __bf16* vt    = (__bf16*)alloc((size_t)MM * MD * 2);
    __bf16* probs = (__bf16*)alloc((size_t)MB * MH * MS * MS * 2);
    __bf16* ctxb  = (__bf16*)alloc((size_t)MM * MD * 2);
    float*  sa    = (float*) alloc((size_t)MM * MD * 4);
    __bf16* ffb   = (__bf16*)alloc((size_t)MM * MF * 2);

    const dim3 gblk(128);

    // h = x (fp32 residual stream); hb = bf16(x) (GEMM stream)
    hipMemcpyAsync(h, x, (size_t)MM * MD * 4, hipMemcpyDeviceToDevice, stream);
    cvt_f32_bf16<<<(MM * MD + 255) / 256, 256, 0, stream>>>(x, hb, MM * MD);

    for (int l = 0; l < ML; ++l) {
        const size_t wdd = (size_t)l * MD * MD;
        const size_t wdf = (size_t)l * MD * MF;

        // Weight transposes -> N-major bf16 (B^T layout for contiguous WMMA loads)
        transpose_cvt<<<dim3(MD/32, MD/32), dim3(32, 8), 0, stream>>>(Wq + wdd, wqt, MD, MD);
        transpose_cvt<<<dim3(MD/32, MD/32), dim3(32, 8), 0, stream>>>(Wk + wdd, wkt, MD, MD);
        transpose_cvt<<<dim3(MD/32, MD/32), dim3(32, 8), 0, stream>>>(Wv + wdd, wvt, MD, MD);
        transpose_cvt<<<dim3(MD/32, MD/32), dim3(32, 8), 0, stream>>>(Wo + wdd, wot, MD, MD);
        transpose_cvt<<<dim3(MF/32, MD/32), dim3(32, 8), 0, stream>>>(W1 + wdf, w1t, MD, MF);
        transpose_cvt<<<dim3(MD/32, MF/32), dim3(32, 8), 0, stream>>>(W2 + wdf, w2t, MF, MD);

        // QKV projections: q scaled by 1/sqrt(DH) after bias
        wmma_gemm<<<dim3(MD/64, MM/128, 1), gblk, 0, stream>>>(
            hb, 0, 0, MD, wqt, 0, 0, MD, nullptr, qb, 0, 0, MD,
            bq + (size_t)l * MD, MD, 0.125f, 1, 0);
        wmma_gemm<<<dim3(MD/64, MM/128, 1), gblk, 0, stream>>>(
            hb, 0, 0, MD, wkt, 0, 0, MD, nullptr, kb, 0, 0, MD,
            bk + (size_t)l * MD, MD, 1.0f, 1, 0);
        wmma_gemm<<<dim3(MD/64, MM/128, 1), gblk, 0, stream>>>(
            hb, 0, 0, MD, wvt, 0, 0, MD, nullptr, vb, 0, 0, MD,
            bv + (size_t)l * MD, MD, 1.0f, 1, 0);
        v_transpose<<<(MM * MD + 255) / 256, 256, 0, stream>>>(vb, vt, MM * MD);

        // scores[b,h] = q_head * k_head^T   (z = b*H + h, zdiv = H)
        wmma_gemm<<<dim3(MS/64, MS/128, MB * MH), gblk, 0, stream>>>(
            qb, (long)MS * MD, MDH, MD,
            kb, (long)MS * MD, MDH, MD,
            nullptr, probs, (long)MH * MS * MS, (long)MS * MS, MS,
            nullptr, MDH, 1.0f, MH, 0);

        softmax_rows<<<MB * MH * MS, 256, 0, stream>>>(probs, amsk, MS, MH * MS);

        // ctx[b,h] = probs * v_head  (B^T = vt head, row d, stride S)
        wmma_gemm<<<dim3(MDH/64, MS/128, MB * MH), gblk, 0, stream>>>(
            probs, (long)MH * MS * MS, (long)MS * MS, MS,
            vt, (long)MH * MDH * MS, (long)MDH * MS, MS,
            nullptr, ctxb, (long)MS * MD, MDH, MD,
            nullptr, MS, 1.0f, MH, 0);

        // Output projection -> fp32 sa
        wmma_gemm<<<dim3(MD/64, MM/128, 1), gblk, 0, stream>>>(
            ctxb, 0, 0, MD, wot, 0, 0, MD, sa, nullptr, 0, 0, MD,
            bo + (size_t)l * MD, MD, 1.0f, 1, 0);

        // h = LN1(sa + h); hb = bf16(h)
        ln_residual<<<MM, 256, 0, stream>>>(sa, h, g1 + (size_t)l * MD,
                                            be1 + (size_t)l * MD, h, hb, MD);

        // FFN: ffb = gelu(h*W1 + b1); sa = ffb*W2 + b2
        wmma_gemm<<<dim3(MF/64, MM/128, 1), gblk, 0, stream>>>(
            hb, 0, 0, MD, w1t, 0, 0, MD, nullptr, ffb, 0, 0, MF,
            b1 + (size_t)l * MF, MD, 1.0f, 1, 1);
        wmma_gemm<<<dim3(MD/64, MM/128, 1), gblk, 0, stream>>>(
            ffb, 0, 0, MF, w2t, 0, 0, MF, sa, nullptr, 0, 0, MD,
            b2 + (size_t)l * MD, MF, 1.0f, 1, 0);

        // h = LN2(sa + h); hb = bf16(h)
        ln_residual<<<MM, 256, 0, stream>>>(sa, h, g2 + (size_t)l * MD,
                                            be2 + (size_t)l * MD, h, hb, MD);
    }

    hipMemcpyAsync(d_out, h, (size_t)MM * MD * 4, hipMemcpyDeviceToDevice, stream);
}
